// CausalLayer_72980084294379
// MI455X (gfx1250) — matrix operand order repaired and checked by reading, not compiled, not run on hardware
//
#include <hip/hip_runtime.h>
#include <hip/hip_bf16.h>

#define Bn 16
#define Ln 2048
#define Hn 768
#define Dn 64
#define LDS_LD 72   // halves; multiple of 8 (16B frag alignment), 36-word lane stride

typedef __attribute__((ext_vector_type(16))) _Float16 v16h;
typedef __attribute__((ext_vector_type(8)))  _Float16 v8h;
typedef __attribute__((ext_vector_type(8)))  float    v8f;

// --- prep: E = ae[x] (f16);  F' = (E @ w) * (i+1)/L (f16), both [b][i][e] ---
// Folding (i+1)/L into F keeps f16 magnitudes bounded (<= raw F) and makes the
// score coefficient fully separable: coef = [(i+1)/L] * [L/(j+1)], where the
// row factor L/(j+1) is applied once to the f32 accumulators in the epilogue.
__global__ __launch_bounds__(256) void prep_kernel(
    const int* __restrict__ x, const float* __restrict__ ae,
    const float* __restrict__ w, _Float16* __restrict__ E16,
    _Float16* __restrict__ F16) {
  __shared__ float sw[64 * 64];
  __shared__ float sE[4][64];
  const int tid = threadIdx.x;
  for (int k = tid; k < 64 * 64; k += 256) sw[k] = w[k];
  const int rg = tid >> 6;
  const int d  = tid & 63;
  const long long row = (long long)blockIdx.x * 4 + rg;   // 0 .. B*L-1
  const int i   = (int)(row & (Ln - 1));
  const int idx = x[row];
  sE[rg][d] = ae[(long long)idx * 64 + d];
  __syncthreads();
  float f = 0.f;
#pragma unroll
  for (int k = 0; k < 64; ++k) f += sE[rg][k] * sw[k * 64 + d];
  const float iscale = (float)(i + 1) * (1.0f / (float)Ln);
  E16[row * 64 + d] = (_Float16)sE[rg][d];
  F16[row * 64 + d] = (_Float16)(f * iscale);
}

// ---- WMMA fragment loads: two contiguous 16B pieces per lane (wave32) ------
__device__ __forceinline__ v16h frag2(const _Float16* r0, const _Float16* r1) {
  v8h lo = *reinterpret_cast<const v8h*>(r0);
  v8h hi = *reinterpret_cast<const v8h*>(r1);
  return __builtin_shufflevector(lo, hi, 0, 1, 2, 3, 4, 5, 6, 7,
                                 8, 9, 10, 11, 12, 13, 14, 15);
}
// A[16xK] 16-bit: lane<16 -> M=lane, K {kh..kh+7, 16+kh..16+kh+7}, kh=0/8
__device__ __forceinline__ v16h load_A16(const _Float16* p) {
  const int lane = threadIdx.x & 31;
  const _Float16* r = p + (lane & 15) * LDS_LD + ((lane & 16) ? 8 : 0);
  return frag2(r, r + 16);
}
// B[32x16] from row-major [n][k] tile: lane<16 -> N=lane, K {kh..kh+7, 16+kh..16+kh+7}, kh=0/8
__device__ __forceinline__ v16h load_B16(const _Float16* p) {
  const int lane = threadIdx.x & 31;
  const _Float16* r = p + (lane & 15) * LDS_LD + ((lane & 16) ? 8 : 0);
  return frag2(r, r + 16);
}

// ---------------- stage F i-tile and transposed V i-tile into LDS -----------
__device__ __forceinline__ void stage_tiles(
    const float* __restrict__ X, const _Float16* __restrict__ F16,
    _Float16* sF, _Float16* sXT, int bb, int i0, int j0, int h0, int tid) {
  {
    const int il = tid >> 2, c = (tid & 3) * 16;
    const uint4* g = reinterpret_cast<const uint4*>(
        F16 + ((long long)bb * Ln + i0 + il) * 64 + c);
    uint4 q0 = g[0], q1 = g[1];
    *reinterpret_cast<uint4*>(&sF[il * LDS_LD + c])     = q0;
    *reinterpret_cast<uint4*>(&sF[il * LDS_LD + c + 8]) = q1;
    if (i0 + 64 <= j0)  // prefetch next F tile (global_prefetch_b8)
      __builtin_prefetch(F16 + ((long long)bb * Ln + i0 + 64 + il) * 64 + c, 0, 3);
  }
  {
    const int hl0 = (tid & 31) * 4;
    const int il0 = (tid >> 5) * 8;
    float vv[8][4];
#pragma unroll
    for (int r8 = 0; r8 < 8; ++r8) {
      float4 t = *reinterpret_cast<const float4*>(
          X + ((long long)bb * Ln + i0 + il0 + r8) * Hn + h0 + hl0);
      vv[r8][0] = t.x; vv[r8][1] = t.y; vv[r8][2] = t.z; vv[r8][3] = t.w;
    }
#pragma unroll
    for (int hh = 0; hh < 4; ++hh) {
      v8h col;
#pragma unroll
      for (int r8 = 0; r8 < 8; ++r8) col[r8] = (_Float16)vv[r8][hh];
      *reinterpret_cast<v8h*>(&sXT[(hl0 + hh) * LDS_LD + il0]) = col;
    }
    if (i0 + 64 <= j0) {
#pragma unroll
      for (int r8 = 0; r8 < 8; r8 += 2)
        __builtin_prefetch(
            X + ((long long)bb * Ln + i0 + 64 + il0 + r8) * Hn + h0 + hl0, 0, 3);
    }
  }
}

// ------- QK^T score tiles; mask only on the peeled diagonal block -----------
template <bool DIAG>
__device__ __forceinline__ void qk_stage(
    const v16h& aE0, const v16h& aE1, const _Float16* sF, _Float16* sS,
    int jb, int nset, int lane, int i0, int j0) {
#pragma unroll
  for (int tt = 0; tt < 2; ++tt) {
    const int ns = nset * 2 + tt;
    v8f c = {};
    v16h b0 = load_B16(sF + ns * 16 * LDS_LD + 0);
    c = __builtin_amdgcn_wmma_f32_16x16x32_f16(false, aE0, false, b0,
                                               (short)0, c, false, false);
    v16h b1 = load_B16(sF + ns * 16 * LDS_LD + 32);
    c = __builtin_amdgcn_wmma_f32_16x16x32_f16(false, aE1, false, b1,
                                               (short)0, c, false, false);
    const int il = ns * 16 + (lane & 15);
#pragma unroll
    for (int r = 0; r < 8; ++r) {
      float v = c[r];
      if (DIAG) {
        const int i = i0 + il, j = j0 + jb + r;
        v = (i < j) ? v : 0.f;   // strict-causal mask
      }
      sS[(jb + r) * LDS_LD + il] = (_Float16)v;
    }
  }
}

// ------- PV: acc += S(64x64) x V(64x128); wave: 4 C tiles, 2 K chunks -------
__device__ __forceinline__ void pv_stage(
    const _Float16* sS, const _Float16* sXT, v8f acc[4], int mT, int nset) {
  v16h a0 = load_A16(sS + mT * 16 * LDS_LD + 0);
  v16h a1 = load_A16(sS + mT * 16 * LDS_LD + 32);
#pragma unroll
  for (int nt = 0; nt < 4; ++nt) {
    const _Float16* bp = sXT + (nset * 64 + nt * 16) * LDS_LD;
    v16h b0 = load_B16(bp + 0);
    v16h b1 = load_B16(bp + 32);
    acc[nt] = __builtin_amdgcn_wmma_f32_16x16x32_f16(
        false, a0, false, b0, (short)0, acc[nt], false, false);
    acc[nt] = __builtin_amdgcn_wmma_f32_16x16x32_f16(
        false, a1, false, b1, (short)0, acc[nt], false, false);
  }
}

// ------------------------- fused causal-bilinear kernel ---------------------
__global__ __launch_bounds__(256) void causal_kernel(
    const float* __restrict__ X, const _Float16* __restrict__ E16,
    const _Float16* __restrict__ F16, float* __restrict__ out) {
  __shared__ _Float16 sE [64 * LDS_LD];   // E j-tile  [j][e]  (A of QK)
  __shared__ _Float16 sF [64 * LDS_LD];   // F i-tile  [i][e]  (B of QK, n=i,k=e)
  __shared__ _Float16 sS [64 * LDS_LD];   // scores    [j][i]  (A of PV)
  __shared__ _Float16 sXT[128 * LDS_LD];  // V^T tile  [h][i]  (B of PV, n=h,k=i)

  const int tid  = threadIdx.x;
  const int lane = tid & 31;
  const int wv   = tid >> 5;              // 8 waves
  const int j0 = blockIdx.x * 64;
  const int h0 = blockIdx.y * 128;
  const int bb = blockIdx.z;

  const int mT   = wv >> 1;               // this wave's M tile (QK and PV)
  const int nset = wv & 1;                // PV: 64-col half of the 128 chunk
  const int jb   = mT * 16 + ((lane & 16) ? 8 : 0);

  // stage E j-tile once
  {
    const int jl = tid >> 2, c = (tid & 3) * 16;
    const uint4* g = reinterpret_cast<const uint4*>(
        E16 + ((long long)bb * Ln + j0 + jl) * 64 + c);
    uint4 q0 = g[0], q1 = g[1];
    *reinterpret_cast<uint4*>(&sE[jl * LDS_LD + c])     = q0;
    *reinterpret_cast<uint4*>(&sE[jl * LDS_LD + c + 8]) = q1;
  }
  __syncthreads();

  // loop-invariant QK A fragments
  const v16h aE0 = load_A16(sE + mT * 16 * LDS_LD + 0);
  const v16h aE1 = load_A16(sE + mT * 16 * LDS_LD + 32);

  v8f acc[4] = {v8f{}, v8f{}, v8f{}, v8f{}};

  // bulk: all i-tiles strictly below the diagonal -> mask-free inner loop
  for (int i0 = 0; i0 < j0; i0 += 64) {
    stage_tiles(X, F16, sF, sXT, bb, i0, j0, h0, tid);
    __syncthreads();
    qk_stage<false>(aE0, aE1, sF, sS, jb, nset, lane, i0, j0);
    __syncthreads();
    pv_stage(sS, sXT, acc, mT, nset);
    __syncthreads();
  }
  // peeled diagonal block: strict-causal mask applied here only
  {
    stage_tiles(X, F16, sF, sXT, bb, j0, j0, h0, tid);
    __syncthreads();
    qk_stage<true>(aE0, aE1, sF, sS, jb, nset, lane, j0, j0);
    __syncthreads();
    pv_stage(sS, sXT, acc, mT, nset);
    __syncthreads();
  }

  // epilogue: out = bert_x + (L/(j+1)) * acc   (row factor applied once)
  float sj[8];
#pragma unroll
  for (int r = 0; r < 8; ++r) sj[r] = (float)Ln / (float)(j0 + jb + r + 1);
#pragma unroll
  for (int nt = 0; nt < 4; ++nt) {
    const int h = h0 + nset * 64 + nt * 16 + (lane & 15);
#pragma unroll
    for (int r = 0; r < 8; ++r) {
      const long long o = ((long long)bb * Ln + j0 + jb + r) * Hn + h;
      out[o] = X[o] + acc[nt][r] * sj[r];
    }
  }
}

extern "C" void kernel_launch(void* const* d_in, const int* in_sizes, int n_in,
                              void* d_out, int out_size, void* d_ws,
                              size_t ws_size, hipStream_t stream) {
  const float* bert_x = (const float*)d_in[0];
  const int*   x      = (const int*)d_in[1];
  const float* ae     = (const float*)d_in[2];
  const float* w      = (const float*)d_in[3];
  float* out = (float*)d_out;

  _Float16* E16 = (_Float16*)d_ws;                 // B*L*64 halves (4 MB)
  _Float16* F16 = E16 + (size_t)Bn * Ln * Dn;      // B*L*64 halves (4 MB)

  prep_kernel<<<(Bn * Ln) / 4, 256, 0, stream>>>(x, ae, w, E16, F16);

  dim3 grid(Ln / 64, Hn / 128, Bn);
  causal_kernel<<<grid, 256, 0, stream>>>(bert_x, E16, F16, out);
}
